// StructuralSimilarityLoss_24464133718338
// MI455X (gfx1250) — matrix operand order repaired and checked by reading, compiled, not run on hardware
//
#include <hip/hip_runtime.h>

// ---------------------------------------------------------------------------
// SSIM fusion loss for MI455X (gfx1250, wave32, WMMA).
// - 256-thread block covers a 64x64 output region (8 waves x two 16x16
//   sub-tiles) with a shared 80x80 f16 halo in LDS -> ~1.34x read
//   amplification (~202 MB total @ 23.3 TB/s ~ 8.7 us bound).
// - Separable 11x11 Gaussian as banded-matrix GEMMs on the matrix cores:
//   per sub-tile and field: 2x vertical + 1x horizontal
//   v_wmma_f32_16x16x32_f16 (8 fields, blur(f)/blur(f^2) shared by both
//   SSIM terms).
// - Wave-private LDS scratch for the C->A layout transpose. Same-wave LDS
//   ops are processed in order (ISA ch.15.5), so only a COMPILER reordering
//   barrier is needed around the transpose -- no s_wait_dscnt drain; the
//   compiler's automatic waitcnt covers the load->WMMA RAW.
// ---------------------------------------------------------------------------

typedef __attribute__((ext_vector_type(16))) _Float16 v16h;
typedef __attribute__((ext_vector_type(8)))  _Float16 v8h;
typedef __attribute__((ext_vector_type(2)))  _Float16 v2h;
typedef __attribute__((ext_vector_type(8)))  float    v8f;

#define IMG      512
#define BT       64                  // output region per block
#define BPD      8                   // blocks per image dim (512/64)
#define NPLANES  48                  // B*C = 16*3
#define NBLK     (NPLANES * BPD * BPD)
#define HALO     80                  // padded halo (BT + 10 -> 80)
#define VALID    74                  // BT + 10
#define HWS      5
#define SSIM_C1  (0.01f * 0.01f)
#define SSIM_C2  (0.03f * 0.03f)

__device__ __forceinline__ v8f wmma16(v16h a, v16h b, v8f c) {
  return __builtin_amdgcn_wmma_f32_16x16x32_f16(false, a, false, b,
                                                (short)0, c, false, false);
}

// Compiler-only memory barrier: same-wave DS ops execute in order in the LDS
// pipeline, so the store->load RAW through the wave-private scratch needs no
// hardware wait -- just forbid IR/scheduler reordering of the LDS accesses.
__device__ __forceinline__ void wave_lds_order() {
  asm volatile("" ::: "memory");
}

__global__ __launch_bounds__(256)
void ssim_tile_kernel(const float* __restrict__ F,
                      const float* __restrict__ S1,
                      const float* __restrict__ S2,
                      float* __restrict__ block_sums)
{
  // Halo stored TRANSPOSED: tileT[field][col][row] (f16), rows/cols 74..79 = 0.
  __shared__ alignas(128) _Float16 tileT[3][HALO][HALO];
  // Per-wave transpose scratch: [wave][row m][32 halfs: pairs (col n, col n+16)].
  __shared__ alignas(128) _Float16 vbh[8][16][32];
  __shared__ float sm[8];

  const int t    = threadIdx.x;
  const int lane = t & 31;
  const int wv   = t >> 5;

  const int bid   = blockIdx.x;
  const int bx    = bid & (BPD - 1);
  const int by    = (bid >> 3) & (BPD - 1);
  const int plane = bid >> 6;

  const int  r0   = by * BT - HWS;       // halo origin (row 0 of tileT)
  const int  c0   = bx * BT - HWS;
  const long pOff = (long)plane * IMG * IMG;

  // Normalization constant of the 1-D Gaussian (sigma = 1.5).
  float gsum = 0.f;
  #pragma unroll
  for (int i = 0; i < 11; ++i) {
    const float d = (float)(i - HWS);
    gsum += __expf(-d * d / 4.5f);
  }
  #define GW(k) (((k) >= 0 && (k) <= 10) \
      ? (__expf(-(float)(((k) - HWS) * ((k) - HWS)) / 4.5f) / gsum) : 0.f)

  // ---- Cooperative zero-padded halo load (row-major global, transposed LDS)
  for (int p = t; p < HALO * HALO; p += 256) {
    const int row = p / HALO;
    const int col = p - row * HALO;
    const int gr  = r0 + row;
    const int gc  = c0 + col;
    float vf = 0.f, v1 = 0.f, v2 = 0.f;
    if (row < VALID && col < VALID && gr >= 0 && gr < IMG && gc >= 0 && gc < IMG) {
      const long idx = pOff + (long)gr * IMG + gc;
      vf = F[idx]; v1 = S1[idx]; v2 = S2[idx];
    }
    tileT[0][col][row] = (_Float16)vf;
    tileT[1][col][row] = (_Float16)v1;
    tileT[2][col][row] = (_Float16)v2;
  }
  __syncthreads();

  const int hi = lane >> 4;
  const int lo = lane & 15;

  // Constant banded Gaussian operands (Av: contiguous K order matching the tileT B loads; Bh: table K order).
  // A (16x32 f16): lane<16 -> M=lane, K 0..15; lane>=16 -> M=lane-16,
  //                K 16..31.   A_v[m,j] = g[j-m].
  // B (32x16 f16): lane<16 -> N=lane, K {0..7,16..23}; lane>=16 -> N=lane-16, K {8..15,24..31}.
  // Horizontal B uses the paired scratch layout: A-slot K holds vblur column
  // colOf(K) = (K>>1) + 16*(K&1), so B_h[K,n] = g[colOf(K)-n].
  v16h Av, Bh;
  #pragma unroll
  for (int i = 0; i < 16; ++i) {
    const int Ka = i + hi * 16;
    Av[i] = (_Float16)GW(Ka - lo);
    const int Kb  = (i < 8 ? i : i + 8) + hi * 8;
    const int cOf = (Kb >> 1) + 16 * (Kb & 1);
    Bh[i] = (_Float16)GW(cOf - lo);
  }

  const v8f zero = {};

  // Separable 2-D blur: vertical (2 WMMAs over the 32-col window), wave-local
  // LDS transpose (packed b32 stores), horizontal (1 WMMA).
  auto blur2d = [&](v16h b0, v16h b1) -> v8f {
    v8f d0 = wmma16(Av, b0, zero);
    v8f d1 = wmma16(Av, b1, zero);
    wave_lds_order();                       // prior A-loads ordered before stores
    #pragma unroll
    for (int i = 0; i < 8; ++i) {           // C layout: VGPR i -> M=i+8*hi, N=lo
      const int m = i + hi * 8;
      v2h pr; pr[0] = (_Float16)d0[i]; pr[1] = (_Float16)d1[i];
      *(v2h*)&vbh[wv][m][2 * lo] = pr;      // ds_store_b32 (fused to 2addr)
    }
    wave_lds_order();                       // stores ordered before A-loads
    const v8h aLo = *(const v8h*)&vbh[wv][lo][hi * 8];       // K 0..7 / 8..15
    const v8h aHi = *(const v8h*)&vbh[wv][lo][16 + hi * 8];  // K 16..23 / 24..31
    const v16h Ah = __builtin_shufflevector(aLo, aHi,
        0, 1, 2, 3, 4, 5, 6, 7, 8, 9, 10, 11, 12, 13, 14, 15);
    return wmma16(Ah, Bh, zero);
  };

  // Each wave handles two 16x16 sub-tiles of the 64x64 region.
  float part = 0.f;
  #pragma unroll
  for (int s = wv * 2; s < wv * 2 + 2; ++s) {
    const int sy = s >> 2;
    const int sx = s & 3;
    const int cb = sx * 16;                 // sub-tile col base in tileT
    const int rb = sy * 16 + hi * 16;       // K row base within a column

    const v16h bf0  = *(const v16h*)&tileT[0][cb + lo     ][rb];
    const v16h bs10 = *(const v16h*)&tileT[1][cb + lo     ][rb];
    const v16h bs20 = *(const v16h*)&tileT[2][cb + lo     ][rb];
    const v16h bf1  = *(const v16h*)&tileT[0][cb + lo + 16][rb];
    const v16h bs11 = *(const v16h*)&tileT[1][cb + lo + 16][rb];
    const v16h bs21 = *(const v16h*)&tileT[2][cb + lo + 16][rb];

    // Eight blurred fields; product-field B-operands are elementwise f16
    // products (same layout) -> v_pk_mul_f16.
    const v8f mF   = blur2d(bf0,         bf1);
    const v8f mS1  = blur2d(bs10,        bs11);
    const v8f mS2  = blur2d(bs20,        bs21);
    const v8f eFF  = blur2d(bf0  * bf0,  bf1  * bf1);
    const v8f eS11 = blur2d(bs10 * bs10, bs11 * bs11);
    const v8f eS22 = blur2d(bs20 * bs20, bs21 * bs21);
    const v8f eFS1 = blur2d(bf0  * bs10, bf1  * bs11);
    const v8f eFS2 = blur2d(bf0  * bs20, bf1  * bs21);

    #pragma unroll
    for (int i = 0; i < 8; ++i) {
      const float mu1 = mF[i], mu2 = mS1[i], mu3 = mS2[i];
      const float m11 = mu1 * mu1, m22 = mu2 * mu2, m33 = mu3 * mu3;
      const float m12 = mu1 * mu2, m13 = mu1 * mu3;
      const float s11 = eFF[i]  - m11;
      const float s22 = eS11[i] - m22;
      const float s33 = eS22[i] - m33;
      const float s12 = eFS1[i] - m12;
      const float s13 = eFS2[i] - m13;
      const float ssimA = ((2.f * m12 + SSIM_C1) * (2.f * s12 + SSIM_C2)) /
                          ((m11 + m22 + SSIM_C1) * (s11 + s22 + SSIM_C2));
      const float ssimB = ((2.f * m13 + SSIM_C1) * (2.f * s13 + SSIM_C2)) /
                          ((m11 + m33 + SSIM_C1) * (s11 + s33 + SSIM_C2));
      part += ssimA + ssimB;
    }
  }

  // Deterministic reduction: wave32 shuffle tree, then fixed-order block sum.
  #pragma unroll
  for (int off = 16; off > 0; off >>= 1)
    part += __shfl_xor(part, off, 32);
  if (lane == 0) sm[wv] = part;
  __syncthreads();
  if (t == 0) {
    float s = 0.f;
    #pragma unroll
    for (int i = 0; i < 8; ++i) s += sm[i];
    block_sums[bid] = s;
  }
  #undef GW
}

// Deterministic single-block final reduction: fixed traversal order.
__global__ __launch_bounds__(256)
void ssim_reduce_kernel(const float* __restrict__ sums, float* __restrict__ out)
{
  __shared__ float smr[256];
  float s = 0.f;
  for (int i = threadIdx.x; i < NBLK; i += 256) s += sums[i];
  smr[threadIdx.x] = s;
  __syncthreads();
  #pragma unroll
  for (int stride = 128; stride > 0; stride >>= 1) {
    if (threadIdx.x < stride) smr[threadIdx.x] += smr[threadIdx.x + stride];
    __syncthreads();
  }
  if (threadIdx.x == 0) {
    const float N = 16.f * 3.f * 512.f * 512.f;   // pixels per mean
    out[0] = 1.f - smr[0] / (2.f * N);            // 1 - (ssim1+ssim2)/2
  }
}

extern "C" void kernel_launch(void* const* d_in, const int* in_sizes, int n_in,
                              void* d_out, int out_size, void* d_ws, size_t ws_size,
                              hipStream_t stream)
{
  (void)in_sizes; (void)n_in; (void)out_size; (void)ws_size;
  const float* F  = (const float*)d_in[0];
  const float* S1 = (const float*)d_in[1];
  const float* S2 = (const float*)d_in[2];
  float* block_sums = (float*)d_ws;         // NBLK * 4 B = 12 KB scratch

  ssim_tile_kernel<<<NBLK, 256, 0, stream>>>(F, S1, S2, block_sums);
  ssim_reduce_kernel<<<1, 256, 0, stream>>>(block_sums, (float*)d_out);
}
